// Attention_39187281608993
// MI455X (gfx1250) — matrix operand order repaired and checked by reading, not compiled, not run on hardware
//
#include <hip/hip_runtime.h>
#include <hip/hip_bf16.h>

typedef _Float16 v16h __attribute__((ext_vector_type(16)));
typedef _Float16 v8h  __attribute__((ext_vector_type(8)));
typedef _Float16 v4h  __attribute__((ext_vector_type(4)));
typedef float    v8f  __attribute__((ext_vector_type(8)));
typedef float    v4f  __attribute__((ext_vector_type(4)));
typedef int      v4i  __attribute__((ext_vector_type(4)));

#define B_   4
#define H_   16
#define S_   2048
#define D_   128
#define BH_  (B_ * H_)
#define TQ   16          // q rows per workgroup
#define NW   8           // waves per workgroup (256 threads, wave32)
#define SROW 2052        // padded LDS row stride in floats (bank de-skew: +4 banks/row)

#define LN_THETA 9.210340371976184f       // ln(10000)
#define INV_SQRT_D 0.08838834764831845f   // 1/sqrt(128)
#define MASK_FILL -60000.0f

// ---------------------------------------------------------------------------
// Prep A: RoPE(K) -> f16, row-major [BH, S, D]. One thread per 8 elements:
// 32B NT load in, 16B store out.
// ---------------------------------------------------------------------------
__global__ void rope_k_f16_kernel(const float* __restrict__ K,
                                  _Float16* __restrict__ K16) {
    size_t idx = (size_t)blockIdx.x * blockDim.x + threadIdx.x;   // BH*S*(D/8)
    if (idx >= (size_t)BH_ * S_ * (D_ / 8)) return;
    int g       = (int)(idx & (D_ / 8 - 1));      // 8-elem group 0..15
    size_t rowi = idx >> 4;                       // bh*S + s
    int s       = (int)(rowi & (S_ - 1));
    int k0      = 8 * g;
    size_t base = rowi * D_ + k0;
    v4f x0 = __builtin_nontemporal_load((const v4f*)(K + base));
    v4f x1 = __builtin_nontemporal_load((const v4f*)(K + base + 4));
    float x[8] = {x0[0], x0[1], x0[2], x0[3], x1[0], x1[1], x1[2], x1[3]};
    v8h o;
    #pragma unroll
    for (int j = 0; j < 8; j += 2) {
        float ang = (float)s * __expf(-(float)(k0 + j) * (LN_THETA / (float)D_));
        float sn, cs;
        __sincosf(ang, &sn, &cs);
        o[j]     = (_Float16)(x[j] * cs - x[j + 1] * sn);
        o[j + 1] = (_Float16)(x[j + 1] * cs + x[j] * sn);
    }
    *(v8h*)(K16 + base) = o;
}

// ---------------------------------------------------------------------------
// Prep B: V (fp32 [BH,S,D]) -> f16 transposed [BH, D, S] via LDS tile.
// ---------------------------------------------------------------------------
__global__ void transpose_v_f16_kernel(const float* __restrict__ V,
                                       _Float16* __restrict__ Vt) {
    __shared__ _Float16 tile[32][34];
    int bh = blockIdx.z;
    int s0 = blockIdx.x * 32;
    int d0 = blockIdx.y * 32;
    int tx = threadIdx.x, ty = threadIdx.y;       // 32 x 8
    const float* vb = V + (size_t)bh * S_ * D_;
    #pragma unroll
    for (int j = ty; j < 32; j += 8)
        tile[j][tx] = (_Float16)__builtin_nontemporal_load(
            vb + (size_t)(s0 + j) * D_ + d0 + tx);
    __syncthreads();
    _Float16* ob = Vt + (size_t)bh * D_ * S_;
    #pragma unroll
    for (int j = ty; j < 32; j += 8)
        ob[(size_t)(d0 + j) * S_ + s0 + tx] = tile[tx][j];
}

// ---------------------------------------------------------------------------
// Attention: one block per (bh, 16-row q tile). 8 waves.
//   GEMM1 (WMMA f16, dual tiles in flight): scores[16 x 2048] -> LDS fp32
//   softmax per row (2 rows/wave, 128-bit vectorized) -> NT fp32 p_attn + f16 P in LDS
//   GEMM2 (WMMA f16, dual accumulator chains): out[16 x 128] = P @ V
// ---------------------------------------------------------------------------
__launch_bounds__(256)
__global__ void attn_kernel(const float* __restrict__ Q,
                            const int* __restrict__ mask,
                            const _Float16* __restrict__ K16,
                            const _Float16* __restrict__ V16t,
                            float* __restrict__ out,
                            float* __restrict__ pattn) {
    __shared__ float smem[TQ * SROW];             // 131,328 B; reused as f16 P in place

    const int tid = threadIdx.x;
    const int w   = tid >> 5;                     // wave id 0..7
    const int l   = tid & 31;                     // lane
    const int lm  = l & 15;
    const int hi  = l >> 4;                       // 0/1: selects k sub-run / M+8
    const int q0  = blockIdx.x * TQ;
    const int bh  = blockIdx.y;

    // ---- Build WMMA A operand: RoPE(Q) * 1/sqrt(D), f16, 4 chunks of K=32 ----
    // A as packed here (contiguous k, as B): lane m<16 holds row m, k=kb..kb+15 contiguous,
    // lanes 16-31 hold k shifted by +16.
    v16h a[4];
    const int qrow = q0 + lm;
    const float* qptr = Q + ((size_t)bh * S_ + qrow) * D_;
    #pragma unroll
    for (int c = 0; c < 4; ++c) {
        const int kb = 32 * c + 16 * hi;
        #pragma unroll
        for (int rr = 0; rr < 2; ++rr) {          // run +0 and +8
            const int kk = kb + 8 * rr;
            v4f y0 = __builtin_nontemporal_load((const v4f*)(qptr + kk));
            v4f y1 = __builtin_nontemporal_load((const v4f*)(qptr + kk + 4));
            float x[8] = {y0[0], y0[1], y0[2], y0[3], y1[0], y1[1], y1[2], y1[3]};
            #pragma unroll
            for (int jj = 0; jj < 8; jj += 2) {   // RoPE pairs stay inside a run
                const int kd = kk + jj;
                float ang = (float)qrow * __expf(-(float)kd * (LN_THETA / (float)D_));
                float sn, cs;
                __sincosf(ang, &sn, &cs);
                a[c][rr * 8 + jj]     = (_Float16)((x[jj] * cs - x[jj + 1] * sn) * INV_SQRT_D);
                a[c][rr * 8 + jj + 1] = (_Float16)((x[jj + 1] * cs + x[jj] * sn) * INV_SQRT_D);
            }
        }
    }

    // ---- GEMM1: scores = (Qr/sqrt(D)) * Kr^T into LDS; 2 key tiles in flight ----
    // B as loaded here (contiguous k, as A): lane n<16 holds column n, k=0..15; lanes 16-31 k=16..31
    // => per lane 16 contiguous f16 from one K row.
    const _Float16* kbase = K16 + (size_t)bh * S_ * D_;
    for (int tt = 0; tt < (S_ / 16) / (2 * NW); ++tt) {   // 8 iterations
        const int n0 = (w + 2 * NW * tt) * 16;
        const int n1 = n0 + NW * 16;
        v8f acc0 = {}, acc1 = {};
        #pragma unroll
        for (int c = 0; c < 4; ++c) {
            const _Float16* bp0 = kbase + (size_t)(n0 + lm) * D_ + 32 * c + 16 * hi;
            const _Float16* bp1 = kbase + (size_t)(n1 + lm) * D_ + 32 * c + 16 * hi;
            v16h b0 = *(const v16h*)bp0;          // 32B aligned contiguous
            v16h b1 = *(const v16h*)bp1;
            acc0 = __builtin_amdgcn_wmma_f32_16x16x32_f16(
                false, a[c], false, b0, (short)0, acc0, false, false);
            acc1 = __builtin_amdgcn_wmma_f32_16x16x32_f16(
                false, a[c], false, b1, (short)0, acc1, false, false);
        }
        // C layout: component r -> (m = r + 8*hi, n = lm)
        #pragma unroll
        for (int r = 0; r < 8; ++r) {
            smem[(r + 8 * hi) * SROW + n0 + lm] = acc0[r];
            smem[(r + 8 * hi) * SROW + n1 + lm] = acc1[r];
        }
    }
    __syncthreads();

    // ---- Softmax: wave w owns rows 2w, 2w+1; 4-wide everywhere ----
    #pragma unroll
    for (int rr = 0; rr < 2; ++rr) {
        const int m  = 2 * w + rr;
        const int qg = q0 + m;
        float* row = smem + m * SROW;
        const int* mrow = mask + (size_t)qg * S_;

        // pass 1: mask-fill in LDS, row max
        float mx = -3.0e38f;
        for (int k4 = l; k4 < S_ / 4; k4 += 32) {
            v4f s = *(v4f*)(row + 4 * k4);
            v4i mv = *(const v4i*)(mrow + 4 * k4);
            #pragma unroll
            for (int j = 0; j < 4; ++j) {
                s[j] = (mv[j] == 0) ? MASK_FILL : s[j];
                mx = fmaxf(mx, s[j]);
            }
            *(v4f*)(row + 4 * k4) = s;
        }
        #pragma unroll
        for (int off = 16; off > 0; off >>= 1)
            mx = fmaxf(mx, __shfl_xor(mx, off, 32));

        // pass 2: exp in LDS, row sum
        float sum = 0.f;
        for (int k4 = l; k4 < S_ / 4; k4 += 32) {
            v4f s = *(v4f*)(row + 4 * k4);
            #pragma unroll
            for (int j = 0; j < 4; ++j) {
                s[j] = __expf(s[j] - mx);
                sum += s[j];
            }
            *(v4f*)(row + 4 * k4) = s;
        }
        #pragma unroll
        for (int off = 16; off > 0; off >>= 1)
            sum += __shfl_xor(sum, off, 32);
        const float inv = 1.0f / sum;

        // pass 3: normalize; NT b128 fp32 p_attn to HBM + packed b64 f16 P in place
        // (write byte range [256i,256i+256) trails read range [512i,512i+512))
        float* prow = pattn + ((size_t)bh * S_ + qg) * S_;
        _Float16* rowh = (_Float16*)row;
        for (int k4 = l; k4 < S_ / 4; k4 += 32) {
            v4f p = *(v4f*)(row + 4 * k4);
            v4h ph;
            #pragma unroll
            for (int j = 0; j < 4; ++j) {
                p[j] *= inv;
                ph[j] = (_Float16)p[j];
            }
            __builtin_nontemporal_store(p, (v4f*)(prow + 4 * k4));
            *(v4h*)(rowh + 4 * k4) = ph;
        }
    }
    __syncthreads();

    // ---- GEMM2: out = P @ V ; wave w owns dims d0..d0+15; even/odd k chains ----
    const int d0 = w * 16;
    const _Float16* vbase = V16t + (size_t)bh * D_ * S_;
    const _Float16* arow  = (const _Float16*)(smem + (size_t)lm * SROW);
    const _Float16* vrow  = vbase + (size_t)(d0 + lm) * S_;
    v8f acc0 = {}, acc1 = {};
    for (int c = 0; c < S_ / 32; c += 2) {
        const int ko0 = 32 * c + 16 * hi;
        const int ko1 = ko0 + 32;
        v8h a0lo = *(const v8h*)(arow + ko0);
        v8h a0hi = *(const v8h*)(arow + ko0 + 8);
        v8h a1lo = *(const v8h*)(arow + ko1);
        v8h a1hi = *(const v8h*)(arow + ko1 + 8);
        v16h av0 = __builtin_shufflevector(a0lo, a0hi,
                                           0, 1, 2, 3, 4, 5, 6, 7,
                                           8, 9, 10, 11, 12, 13, 14, 15);
        v16h av1 = __builtin_shufflevector(a1lo, a1hi,
                                           0, 1, 2, 3, 4, 5, 6, 7,
                                           8, 9, 10, 11, 12, 13, 14, 15);
        v16h bv0 = *(const v16h*)(vrow + 32 * c + 16 * hi);
        v16h bv1 = *(const v16h*)(vrow + 32 * (c + 1) + 16 * hi);
        acc0 = __builtin_amdgcn_wmma_f32_16x16x32_f16(
            false, av0, false, bv0, (short)0, acc0, false, false);
        acc1 = __builtin_amdgcn_wmma_f32_16x16x32_f16(
            false, av1, false, bv1, (short)0, acc1, false, false);
    }
    float* obase = out + ((size_t)bh * S_ + q0) * D_ + d0 + lm;
    #pragma unroll
    for (int r = 0; r < 8; ++r)
        obase[(size_t)(r + 8 * hi) * D_] = acc0[r] + acc1[r];
}

// ---------------------------------------------------------------------------
extern "C" void kernel_launch(void* const* d_in, const int* in_sizes, int n_in,
                              void* d_out, int out_size, void* d_ws, size_t ws_size,
                              hipStream_t stream) {
    const float* Q    = (const float*)d_in[0];
    const float* K    = (const float*)d_in[1];
    const float* V    = (const float*)d_in[2];
    const int*   mask = (const int*)d_in[3];

    float* out   = (float*)d_out;                       // [BH, S, D]
    float* pattn = out + (size_t)BH_ * S_ * D_;         // [BH, S, S]

    _Float16* K16  = (_Float16*)d_ws;                   // 33.5 MB
    _Float16* V16t = K16 + (size_t)BH_ * S_ * D_;       // 33.5 MB

    const size_t ngroups = (size_t)BH_ * S_ * (D_ / 8);
    rope_k_f16_kernel<<<(unsigned)((ngroups + 255) / 256), 256, 0, stream>>>(K, K16);

    dim3 tgrid(S_ / 32, D_ / 32, BH_);
    transpose_v_f16_kernel<<<tgrid, dim3(32, 8, 1), 0, stream>>>(V, V16t);

    attn_kernel<<<dim3(S_ / TQ, BH_, 1), 256, 0, stream>>>(Q, mask, K16, V16t,
                                                           out, pattn);
}
